// WanAttentionBlock_86998857547916
// MI455X (gfx1250) — matrix operand order repaired and checked by reading, not compiled, not run on hardware
//
#include <hip/hip_runtime.h>
#include <hip/hip_bf16.h>
#include <stdint.h>

// ---------------- problem constants (from reference) ----------------
#define DIMC   1536
#define HEADSC 12
#define HDC    128
#define FFNC   8960
#define STOK   3456          // 8*18*24
#define T5C    512
#define IMGC   257
#define IMGPC  288           // 257 padded to multiple of 32
#define L2C    769
#define FF     8
#define HH     18
#define WW     24
#define EPSF   1e-6f
#define QSCALE 0.08838834764831845f   // 1/sqrt(128)

typedef __bf16 bf16;
typedef __attribute__((ext_vector_type(8)))  bf16         bf16x8;
typedef __attribute__((ext_vector_type(16))) bf16         bf16x16;
typedef __attribute__((ext_vector_type(8)))  float        f32x8;
typedef __attribute__((ext_vector_type(8)))  unsigned int u32x8;

// ---------------- helpers ----------------
__device__ __forceinline__ bf16 to_bf16(float f) {
  union { float f; uint32_t u; } v; v.f = f;
  uint32_t r = v.u + 0x7FFFu + ((v.u >> 16) & 1u);   // round-to-nearest-even
  unsigned short h = (unsigned short)(r >> 16);
  return __builtin_bit_cast(bf16, h);
}
__device__ __forceinline__ f32x8 zero_f32x8() {
  u32x8 z = (u32x8)0u; return __builtin_bit_cast(f32x8, z);
}
__device__ __forceinline__ bf16x16 zero_bf16x16() {
  u32x8 z = (u32x8)0u; return __builtin_bit_cast(bf16x16, z);
}
__device__ __forceinline__ bf16x16 join16(bf16x8 lo, bf16x8 hi) {
  union U { bf16x16 v; bf16x8 h[2]; } u;
  u.h[0] = lo; u.h[1] = hi; return u.v;
}
__device__ __forceinline__ f32x8 wmma_bf16(bf16x16 a, bf16x16 b, f32x8 c) {
  // D = A(16x32 bf16) x B(32x16 bf16) + C(16x16 f32)  -> v_wmma_f32_16x16x32_bf16
  return __builtin_amdgcn_wmma_f32_16x16x32_bf16(false, a, false, b, (short)0, c, false, false);
}
__device__ __forceinline__ float gelu_tanh_f(float z) {
  float z3 = z * z * z;
  return 0.5f * z * (1.0f + tanhf(0.7978845608028654f * (z + 0.044715f * z3)));
}

// ---------------- small elementwise kernels ----------------
__global__ void addmod_kernel(const float* __restrict__ mod, const float* __restrict__ e,
                              float* __restrict__ e6) {
  int i = blockIdx.x * 256 + threadIdx.x;
  if (i < 6 * DIMC) e6[i] = mod[i] + e[i];
}

// y = ln(x) * (add1 + scale[col]) + shift[col]   (row-wise LN over D)
__global__ __launch_bounds__(256)
void ln_kernel(const float* __restrict__ x, float* __restrict__ outf, bf16* __restrict__ outb,
               const float* __restrict__ scale, const float* __restrict__ shift,
               float add1, int D) {
  int row = blockIdx.x;
  const float* xr = x + (size_t)row * D;
  float s = 0.f, s2 = 0.f;
  for (int i = threadIdx.x; i < D; i += 256) { float v = xr[i]; s += v; s2 += v * v; }
#pragma unroll
  for (int off = 16; off; off >>= 1) { s += __shfl_down(s, off, 32); s2 += __shfl_down(s2, off, 32); }
  __shared__ float rs[8], rs2[8];
  __shared__ float mean_sh, rstd_sh;
  int wid = threadIdx.x >> 5, lane = threadIdx.x & 31;
  if (lane == 0) { rs[wid] = s; rs2[wid] = s2; }
  __syncthreads();
  if (threadIdx.x == 0) {
    float ts = 0.f, ts2 = 0.f;
    for (int i = 0; i < 8; ++i) { ts += rs[i]; ts2 += rs2[i]; }
    float mean = ts / (float)D;
    float var  = ts2 / (float)D - mean * mean;
    mean_sh = mean; rstd_sh = rsqrtf(var + EPSF);
  }
  __syncthreads();
  float mean = mean_sh, rstd = rstd_sh;
  for (int i = threadIdx.x; i < D; i += 256) {
    float yn = (xr[i] - mean) * rstd;
    float y  = yn * (add1 + scale[i]) + shift[i];
    if (outf) outf[(size_t)row * D + i] = y;
    if (outb) outb[(size_t)row * D + i] = to_bf16(y);
  }
}

// x := x * rsqrt(mean(x^2)+eps) * w   (in place, row-wise over D)
__global__ __launch_bounds__(256)
void rms_kernel(float* __restrict__ x, const float* __restrict__ w, int D) {
  int row = blockIdx.x;
  float* xr = x + (size_t)row * D;
  float s2 = 0.f;
  for (int i = threadIdx.x; i < D; i += 256) { float v = xr[i]; s2 += v * v; }
#pragma unroll
  for (int off = 16; off; off >>= 1) s2 += __shfl_down(s2, off, 32);
  __shared__ float rs2[8];
  __shared__ float rr_sh;
  int wid = threadIdx.x >> 5, lane = threadIdx.x & 31;
  if (lane == 0) rs2[wid] = s2;
  __syncthreads();
  if (threadIdx.x == 0) {
    float ts2 = 0.f;
    for (int i = 0; i < 8; ++i) ts2 += rs2[i];
    rr_sh = rsqrtf(ts2 / (float)D + EPSF);
  }
  __syncthreads();
  float r = rr_sh;
  for (int i = threadIdx.x; i < D; i += 256) xr[i] = xr[i] * r * w[i];
}

// RoPE + bf16 convert (with optional scale) for one (token, head); 64 threads = 64 pairs
__global__ void rope_cvt_kernel(const float* __restrict__ x, const float* __restrict__ freqs,
                                bf16* __restrict__ outb, float scalev) {
  int tok = blockIdx.x, head = blockIdx.y, t = threadIdx.x;     // t in [0,64)
  int f = tok / (HH * WW);
  int rem = tok % (HH * WW);
  int h = rem / WW;
  int w = rem % WW;
  int src = (t < 22) ? f : (t < 43) ? h : w;                    // c0=22, c1=21
  float th = freqs[src * (HDC / 2) + t];
  float cs = __cosf(th), sn = __sinf(th);
  size_t base = (size_t)tok * DIMC + head * HDC + 2 * t;
  float re = x[base], im = x[base + 1];
  outb[base]     = to_bf16((re * cs - im * sn) * scalev);
  outb[base + 1] = to_bf16((re * sn + im * cs) * scalev);
}

// bf16 convert with scale and zero-padding of rows >= rows_valid
__global__ void cvt_pad_kernel(const float* __restrict__ in, bf16* __restrict__ out,
                               int rows_out, int cols, int rows_valid, float scalev) {
  size_t idx = (size_t)blockIdx.x * 256 + threadIdx.x;
  size_t total = (size_t)rows_out * cols;
  if (idx >= total) return;
  int r = (int)(idx / cols);
  float v = (r < rows_valid) ? in[idx] * scalev : 0.f;
  out[idx] = to_bf16(v);
}

// vt[h][d][t] = bf16(v[t][h*HD+d]) with zero pad t >= rows_valid
__global__ void vtrans_kernel(const float* __restrict__ v, bf16* __restrict__ vt,
                              int Lpad, int rows_valid) {
  size_t idx = (size_t)blockIdx.x * 256 + threadIdx.x;
  size_t total = (size_t)HEADSC * HDC * Lpad;
  if (idx >= total) return;
  int t = (int)(idx % Lpad);
  int rem = (int)(idx / Lpad);
  int d = rem % HDC;
  int h = rem / HDC;
  float val = (t < rows_valid) ? v[(size_t)t * DIMC + h * HDC + d] : 0.f;
  vt[idx] = to_bf16(val);
}

// Bt[n][k] = bf16(W[k][n]) for W (K x N) -> weight transpose+convert
__global__ void wt_cvt_kernel(const float* __restrict__ W, bf16* __restrict__ Bt, int K, int N) {
  size_t idx = (size_t)blockIdx.x * 256 + threadIdx.x;
  size_t total = (size_t)K * N;
  if (idx >= total) return;
  int n = (int)(idx / K);
  int k = (int)(idx % K);
  Bt[idx] = to_bf16(W[(size_t)k * N + n]);
}

__global__ void resgate_kernel(const float* __restrict__ x, const float* __restrict__ y,
                               const float* __restrict__ erow, float* __restrict__ out, int total) {
  int idx = blockIdx.x * 256 + threadIdx.x;
  if (idx >= total) return;
  int col = idx % DIMC;
  out[idx] = x[idx] + y[idx] * erow[col];
}

__global__ void resadd_kernel(const float* __restrict__ x, const float* __restrict__ y,
                              float* __restrict__ out, int total) {
  int idx = blockIdx.x * 256 + threadIdx.x;
  if (idx >= total) return;
  out[idx] = x[idx] + y[idx];
}

// ---------------- WMMA GEMM: C[M,N] = A[M,K] @ Bt[N,K]^T + bias ----------------
// A row-major bf16, Bt = B transposed (N-major) bf16 so both operands load
// contiguously in the CDNA5 A/B VGPR layouts. 8 waves/block arranged 2x4:
// workgroup tile 64x256; each wave computes 32x64 = 2 A-tiles x 4 B-tiles
// (8 accumulators) so every B load feeds two WMMAs. K streamed in steps of 32.
// Wave-uniform fast path avoids per-iteration EXEC churn when the M-tile is
// fully in bounds (all GEMMs except the 257-row image-context tail tiles).
__global__ __launch_bounds__(256)
void gemm_bf16_wmma(const bf16* __restrict__ A, const bf16* __restrict__ Bt,
                    const float* __restrict__ bias,
                    float* __restrict__ Cf, bf16* __restrict__ Cb,
                    int M, int N, int K, int fuse_gelu) {
  const int lane = threadIdx.x & 31;
  const int wid  = threadIdx.x >> 5;
  const int m0   = blockIdx.y * 64 + (wid >> 2) * 32;
  const int n0   = blockIdx.x * 256 + (wid & 3) * 64;
  const int l15  = lane & 15;
  const int hi   = lane >> 4;

  f32x8 acc[2][4];
#pragma unroll
  for (int u = 0; u < 2; ++u)
#pragma unroll
    for (int t = 0; t < 4; ++t) acc[u][t] = zero_f32x8();

  // A 16x32 layout: lanes 0-15 elems {K0..7, K16..23}; lanes 16-31 elems {K8..15, K24..31}
  const bf16* arow0 = A + (size_t)(m0 + l15) * K + hi * 8;
  const bf16* arow1 = arow0 + (size_t)16 * K;
  // B 32x16 layout: lanes 0-15 hold K0..7, K16..23 of column (lane&15); lanes 16-31 hold K8..15, K24..31
  const bf16* brow  = Bt + (size_t)(n0 + l15) * K + hi * 8;

  if (m0 + 32 <= M) {
    // -------- fast path: whole 32-row tile in bounds (wave-uniform) --------
    for (int kk = 0; kk < K; kk += 32) {
      __builtin_prefetch(arow0 + kk + 64, 0, 0);   // global_prefetch_b8 next K-slice
      bf16x16 a0 = join16(*(const bf16x8*)(arow0 + kk), *(const bf16x8*)(arow0 + kk + 16));
      bf16x16 a1 = join16(*(const bf16x8*)(arow1 + kk), *(const bf16x8*)(arow1 + kk + 16));
#pragma unroll
      for (int t = 0; t < 4; ++t) {
        const bf16* bp = brow + (size_t)(t * 16) * K + kk;
        bf16x16 b = join16(*(const bf16x8*)bp, *(const bf16x8*)(bp + 16));
        acc[0][t] = wmma_bf16(a0, b, acc[0][t]);
        acc[1][t] = wmma_bf16(a1, b, acc[1][t]);
      }
    }
  } else {
    // -------- guarded path: 257-row context tail --------
    const bool v0 = (m0 + l15) < M;
    const bool v1 = (m0 + 16 + l15) < M;
    for (int kk = 0; kk < K; kk += 32) {
      bf16x16 a0 = zero_bf16x16(), a1 = zero_bf16x16();
      if (v0) a0 = join16(*(const bf16x8*)(arow0 + kk), *(const bf16x8*)(arow0 + kk + 16));
      if (v1) a1 = join16(*(const bf16x8*)(arow1 + kk), *(const bf16x8*)(arow1 + kk + 16));
#pragma unroll
      for (int t = 0; t < 4; ++t) {
        const bf16* bp = brow + (size_t)(t * 16) * K + kk;
        bf16x16 b = join16(*(const bf16x8*)bp, *(const bf16x8*)(bp + 16));
        acc[0][t] = wmma_bf16(a0, b, acc[0][t]);
        acc[1][t] = wmma_bf16(a1, b, acc[1][t]);
      }
    }
  }

  // C layout: VGPR r -> row r (+8 for upper half-wave), N = lane&15
#pragma unroll
  for (int u = 0; u < 2; ++u) {
    const int orow_base = m0 + u * 16 + hi * 8;
#pragma unroll
    for (int t = 0; t < 4; ++t) {
      int col = n0 + t * 16 + l15;
      float bv = bias ? bias[col] : 0.f;
#pragma unroll
      for (int r = 0; r < 8; ++r) {
        int row = orow_base + r;
        if (row < M) {
          float v = acc[u][t][r] + bv;
          if (fuse_gelu) v = gelu_tanh_f(v);
          if (Cf) Cf[(size_t)row * N + col] = v;
          if (Cb) Cb[(size_t)row * N + col] = to_bf16(v);
        }
      }
    }
  }
}

// ---------------- flash attention with WMMA ----------------
// One wave per (head, 16-query tile); 4 waves per block. Q pre-scaled by 1/sqrt(HD).
// Kb row-major [token][DIM] is directly the B matrix of Q*K^T.
// Vt is per-head d-major [HEADS][HD][Lpad] so P*V B-tiles load contiguously.
// P is re-laid out C->A through a private 1KB LDS slab per wave (s_wait_dscnt).
__global__ __launch_bounds__(128)
void flash_attn_wmma(const bf16* __restrict__ Q, const bf16* __restrict__ Kb,
                     const bf16* __restrict__ Vt, float* __restrict__ O,
                     int Sq, int Lk, int Lpad, int accumulate) {
  __shared__ __align__(16) bf16 plds[4][16 * 32];
  const int lane = threadIdx.x & 31;
  const int wid  = threadIdx.x >> 5;
  const int head = blockIdx.y;
  const int qt   = blockIdx.x * 4 + wid;
  if (qt * 16 >= Sq) return;          // wave-uniform; no barriers used below
  const int m0  = qt * 16;
  const int l15 = lane & 15;
  const int hi  = lane >> 4;

  bf16x16 qa[4];
  {
    const bf16* qrow = Q + (size_t)(m0 + l15) * DIMC + head * HDC + hi * 8;
#pragma unroll
    for (int c = 0; c < 4; ++c)
      qa[c] = join16(*(const bf16x8*)(qrow + c * 32), *(const bf16x8*)(qrow + c * 32 + 16));
  }

  const bf16* vhead = Vt + (size_t)head * HDC * Lpad;
  float mrow[8], lrow[8], sc[8];
  f32x8 acc[8];
#pragma unroll
  for (int r = 0; r < 8; ++r) { mrow[r] = -1e30f; lrow[r] = 0.f; }
#pragma unroll
  for (int n = 0; n < 8; ++n) acc[n] = zero_f32x8();

  const int ntiles = Lpad / 32;
  for (int j = 0; j < ntiles; ++j) {
    f32x8 s0 = zero_f32x8(), s1 = zero_f32x8();
#pragma unroll
    for (int c = 0; c < 4; ++c) {
      const bf16* kp0 = Kb + (size_t)(j * 32 + l15) * DIMC + head * HDC + c * 32 + hi * 8;
      bf16x16 b0 = join16(*(const bf16x8*)kp0, *(const bf16x8*)(kp0 + 16));
      s0 = wmma_bf16(qa[c], b0, s0);
      const bf16* kp1 = kp0 + (size_t)16 * DIMC;
      bf16x16 b1 = join16(*(const bf16x8*)kp1, *(const bf16x8*)(kp1 + 16));
      s1 = wmma_bf16(qa[c], b1, s1);
    }
    // mask padded keys (per-lane column)
    {
      int col0 = j * 32 + l15;
      if (col0 >= Lk) {
#pragma unroll
        for (int r = 0; r < 8; ++r) s0[r] = -1e30f;
      }
      if (col0 + 16 >= Lk) {
#pragma unroll
        for (int r = 0; r < 8; ++r) s1[r] = -1e30f;
      }
    }
    // online softmax: rows live across 16 lanes of each half-wave
#pragma unroll
    for (int r = 0; r < 8; ++r) {
      float v = fmaxf(s0[r], s1[r]);
#pragma unroll
      for (int off = 1; off < 16; off <<= 1) v = fmaxf(v, __shfl_xor(v, off, 32));
      float mn = fmaxf(mrow[r], v);
      float e0 = __expf(s0[r] - mn);
      float e1 = __expf(s1[r] - mn);
      s0[r] = e0; s1[r] = e1;
      float ps = e0 + e1;
#pragma unroll
      for (int off = 1; off < 16; off <<= 1) ps += __shfl_xor(ps, off, 32);
      sc[r]   = __expf(mrow[r] - mn);
      lrow[r] = lrow[r] * sc[r] + ps;
      mrow[r] = mn;
    }
#pragma unroll
    for (int n = 0; n < 8; ++n)
#pragma unroll
      for (int r = 0; r < 8; ++r) acc[n][r] *= sc[r];

    // transpose P (C layout) -> A layout through per-wave LDS slab
    bf16* pl = &plds[wid][0];
    int prow = hi * 8;
#pragma unroll
    for (int r = 0; r < 8; ++r) {
      pl[(prow + r) * 32 + l15]      = to_bf16(s0[r]);
      pl[(prow + r) * 32 + 16 + l15] = to_bf16(s1[r]);
    }
    asm volatile("s_wait_dscnt 0x0" ::: "memory");
    const bf16* pr = pl + l15 * 32 + hi * 8;
    bf16x16 pa = join16(*(const bf16x8*)pr, *(const bf16x8*)(pr + 16));
#pragma unroll
    for (int n = 0; n < 8; ++n) {
      const bf16* vp = vhead + (size_t)(n * 16 + l15) * Lpad + j * 32 + hi * 8;
      bf16x16 bv = join16(*(const bf16x8*)vp, *(const bf16x8*)(vp + 16));
      acc[n] = wmma_bf16(pa, bv, acc[n]);
    }
    asm volatile("" ::: "memory");   // keep next iter's LDS stores after these reads
  }

#pragma unroll
  for (int n = 0; n < 8; ++n) {
    int col = head * HDC + n * 16 + l15;
#pragma unroll
    for (int r = 0; r < 8; ++r) {
      int row = m0 + hi * 8 + r;
      float v = acc[n][r] / lrow[r];
      size_t idx = (size_t)row * DIMC + col;
      if (accumulate) O[idx] += v; else O[idx] = v;
    }
  }
}

// ---------------- host-side orchestration ----------------
extern "C" void kernel_launch(void* const* d_in, const int* in_sizes, int n_in,
                              void* d_out, int out_size, void* d_ws, size_t ws_size,
                              hipStream_t stream) {
  (void)in_sizes; (void)n_in; (void)out_size; (void)ws_size;

  const float* x          = (const float*)d_in[0];
  const float* e          = (const float*)d_in[1];
  const float* context    = (const float*)d_in[2];
  const float* freqs      = (const float*)d_in[3];
  const float* modulation = (const float*)d_in[4];
  const float* sa_qw = (const float*)d_in[5];  const float* sa_qb = (const float*)d_in[6];
  const float* sa_kw = (const float*)d_in[7];  const float* sa_kb = (const float*)d_in[8];
  const float* sa_vw = (const float*)d_in[9];  const float* sa_vb = (const float*)d_in[10];
  const float* sa_ow = (const float*)d_in[11]; const float* sa_ob = (const float*)d_in[12];
  const float* sa_nq = (const float*)d_in[13]; const float* sa_nk = (const float*)d_in[14];
  const float* norm3_w = (const float*)d_in[15]; const float* norm3_b = (const float*)d_in[16];
  const float* ca_qw = (const float*)d_in[17]; const float* ca_qb = (const float*)d_in[18];
  const float* ca_kw = (const float*)d_in[19]; const float* ca_kb = (const float*)d_in[20];
  const float* ca_vw = (const float*)d_in[21]; const float* ca_vb = (const float*)d_in[22];
  const float* ca_kiw = (const float*)d_in[23]; const float* ca_kib = (const float*)d_in[24];
  const float* ca_viw = (const float*)d_in[25]; const float* ca_vib = (const float*)d_in[26];
  const float* ca_ow = (const float*)d_in[27]; const float* ca_ob = (const float*)d_in[28];
  const float* ca_nq = (const float*)d_in[29]; const float* ca_nk = (const float*)d_in[30];
  const float* ca_nki = (const float*)d_in[31];
  const float* ffn_w1 = (const float*)d_in[32]; const float* ffn_b1 = (const float*)d_in[33];
  const float* ffn_w2 = (const float*)d_in[34]; const float* ffn_b2 = (const float*)d_in[35];
  float* out = (float*)d_out;

  // ---- workspace bump allocator (deterministic layout every call) ----
  char* ws = (char*)d_ws;
  size_t off = 0;
  auto alloc = [&](size_t bytes) -> void* {
    void* p = ws + off;
    off = (off + bytes + 255) & ~(size_t)255;
    return p;
  };
  const size_t SD = (size_t)STOK * DIMC;

  float* e6   = (float*)alloc(6 * DIMC * 4);
  float* xs   = (float*)alloc(SD * 4);     // reused later as xc (f32)
  float* qf   = (float*)alloc(SD * 4);     // reused for cross-attn q
  float* kf   = (float*)alloc(SD * 4);
  float* vf   = (float*)alloc(SD * 4);
  float* attn = (float*)alloc(SD * 4);     // flash output (self, then cross accum)
  float* yb   = (float*)alloc(SD * 4);     // projection outputs
  float* x1   = (float*)alloc(SD * 4);
  float* x2   = (float*)alloc(SD * 4);
  float* ckf  = (float*)alloc((size_t)T5C * DIMC * 4);
  float* cvf  = (float*)alloc((size_t)T5C * DIMC * 4);
  float* ckif = (float*)alloc((size_t)IMGC * DIMC * 4);
  float* cvif = (float*)alloc((size_t)IMGC * DIMC * 4);

  bf16* w_qT  = (bf16*)alloc((size_t)DIMC * DIMC * 2);
  bf16* w_kT  = (bf16*)alloc((size_t)DIMC * DIMC * 2);
  bf16* w_vT  = (bf16*)alloc((size_t)DIMC * DIMC * 2);
  bf16* w_oT  = (bf16*)alloc((size_t)DIMC * DIMC * 2);
  bf16* cw_qT = (bf16*)alloc((size_t)DIMC * DIMC * 2);
  bf16* cw_kT = (bf16*)alloc((size_t)DIMC * DIMC * 2);
  bf16* cw_vT = (bf16*)alloc((size_t)DIMC * DIMC * 2);
  bf16* cw_kiT = (bf16*)alloc((size_t)DIMC * DIMC * 2);
  bf16* cw_viT = (bf16*)alloc((size_t)DIMC * DIMC * 2);
  bf16* cw_oT = (bf16*)alloc((size_t)DIMC * DIMC * 2);
  bf16* w1T   = (bf16*)alloc((size_t)FFNC * DIMC * 2);   // [N=FFN][K=DIM]
  bf16* w2T   = (bf16*)alloc((size_t)DIMC * FFNC * 2);   // [N=DIM][K=FFN]

  bf16* xs_b   = (bf16*)alloc(SD * 2);    // reused as xc_b and z_b
  bf16* q_b    = (bf16*)alloc(SD * 2);    // reused for cross q
  bf16* k_b    = (bf16*)alloc(SD * 2);
  bf16* vt_b   = (bf16*)alloc(SD * 2);    // [HEADS][HD][STOK]
  bf16* attn_b = (bf16*)alloc(SD * 2);
  bf16* ctx_b  = (bf16*)alloc((size_t)L2C * DIMC * 2);
  bf16* ck_b   = (bf16*)alloc((size_t)T5C * DIMC * 2);
  bf16* cvt_b  = (bf16*)alloc((size_t)T5C * DIMC * 2);   // [HEADS][HD][T5]
  bf16* cki_b  = (bf16*)alloc((size_t)IMGPC * DIMC * 2); // zero-padded rows
  bf16* cvit_b = (bf16*)alloc((size_t)IMGPC * DIMC * 2); // [HEADS][HD][IMGP]
  bf16* h_b    = (bf16*)alloc((size_t)STOK * FFNC * 2);

  auto blocks = [](size_t n) { return (unsigned)((n + 255) / 256); };
  auto wt = [&](const float* W, bf16* Bt, int K, int N) {
    wt_cvt_kernel<<<blocks((size_t)K * N), 256, 0, stream>>>(W, Bt, K, N);
  };
  auto gemm = [&](const bf16* A, const bf16* Bt, const float* bias, float* Cf, bf16* Cb,
                  int M, int N, int K, int fgelu) {
    dim3 g(N / 256, (M + 63) / 64);
    gemm_bf16_wmma<<<g, 256, 0, stream>>>(A, Bt, bias, Cf, Cb, M, N, K, fgelu);
  };
  auto flash = [&](const bf16* Q, const bf16* K, const bf16* Vt, float* O,
                   int Lk, int Lpad, int accum) {
    dim3 g((STOK / 16 + 3) / 4, HEADSC);
    flash_attn_wmma<<<g, 128, 0, stream>>>(Q, K, Vt, O, STOK, Lk, Lpad, accum);
  };

  // ---- weight convert+transpose (bf16, N-major) ----
  wt(sa_qw, w_qT, DIMC, DIMC);   wt(sa_kw, w_kT, DIMC, DIMC);
  wt(sa_vw, w_vT, DIMC, DIMC);   wt(sa_ow, w_oT, DIMC, DIMC);
  wt(ca_qw, cw_qT, DIMC, DIMC);  wt(ca_kw, cw_kT, DIMC, DIMC);
  wt(ca_vw, cw_vT, DIMC, DIMC);  wt(ca_kiw, cw_kiT, DIMC, DIMC);
  wt(ca_viw, cw_viT, DIMC, DIMC); wt(ca_ow, cw_oT, DIMC, DIMC);
  wt(ffn_w1, w1T, DIMC, FFNC);   wt(ffn_w2, w2T, FFNC, DIMC);

  // ---- modulation + self-attn input ----
  addmod_kernel<<<blocks(6 * DIMC), 256, 0, stream>>>(modulation, e, e6);
  ln_kernel<<<STOK, 256, 0, stream>>>(x, xs, xs_b, e6 + 1 * DIMC, e6 + 0 * DIMC, 1.0f, DIMC);

  gemm(xs_b, w_qT, sa_qb, qf, nullptr, STOK, DIMC, DIMC, 0);
  gemm(xs_b, w_kT, sa_kb, kf, nullptr, STOK, DIMC, DIMC, 0);
  gemm(xs_b, w_vT, sa_vb, vf, nullptr, STOK, DIMC, DIMC, 0);
  rms_kernel<<<STOK, 256, 0, stream>>>(qf, sa_nq, DIMC);
  rms_kernel<<<STOK, 256, 0, stream>>>(kf, sa_nk, DIMC);
  rope_cvt_kernel<<<dim3(STOK, HEADSC), 64, 0, stream>>>(qf, freqs, q_b, QSCALE);
  rope_cvt_kernel<<<dim3(STOK, HEADSC), 64, 0, stream>>>(kf, freqs, k_b, 1.0f);
  vtrans_kernel<<<blocks(SD), 256, 0, stream>>>(vf, vt_b, STOK, STOK);

  flash(q_b, k_b, vt_b, attn, STOK, STOK, 0);
  cvt_pad_kernel<<<blocks(SD), 256, 0, stream>>>(attn, attn_b, STOK, DIMC, STOK, 1.0f);
  gemm(attn_b, w_oT, sa_ob, yb, nullptr, STOK, DIMC, DIMC, 0);
  resgate_kernel<<<blocks(SD), 256, 0, stream>>>(x, yb, e6 + 2 * DIMC, x1, (int)SD);

  // ---- cross attention ----
  float* xc = xs;           // reuse
  bf16*  xc_b = xs_b;       // reuse
  ln_kernel<<<STOK, 256, 0, stream>>>(x1, xc, xc_b, norm3_w, norm3_b, 0.0f, DIMC);
  cvt_pad_kernel<<<blocks((size_t)L2C * DIMC), 256, 0, stream>>>(context, ctx_b, L2C, DIMC, L2C, 1.0f);
  const bf16* ctx_img_b = ctx_b;                              // first 257 rows
  const bf16* ctx_txt_b = ctx_b + (size_t)IMGC * DIMC;        // last 512 rows

  gemm(xc_b, cw_qT, ca_qb, qf, nullptr, STOK, DIMC, DIMC, 0);
  gemm(ctx_txt_b, cw_kT, ca_kb, ckf, nullptr, T5C, DIMC, DIMC, 0);
  gemm(ctx_txt_b, cw_vT, ca_vb, cvf, nullptr, T5C, DIMC, DIMC, 0);
  gemm(ctx_img_b, cw_kiT, ca_kib, ckif, nullptr, IMGC, DIMC, DIMC, 0);
  gemm(ctx_img_b, cw_viT, ca_vib, cvif, nullptr, IMGC, DIMC, DIMC, 0);
  rms_kernel<<<STOK, 256, 0, stream>>>(qf, ca_nq, DIMC);
  rms_kernel<<<T5C, 256, 0, stream>>>(ckf, ca_nk, DIMC);
  rms_kernel<<<IMGC, 256, 0, stream>>>(ckif, ca_nki, DIMC);

  cvt_pad_kernel<<<blocks(SD), 256, 0, stream>>>(qf, q_b, STOK, DIMC, STOK, QSCALE);
  cvt_pad_kernel<<<blocks((size_t)T5C * DIMC), 256, 0, stream>>>(ckf, ck_b, T5C, DIMC, T5C, 1.0f);
  vtrans_kernel<<<blocks((size_t)HEADSC * HDC * T5C), 256, 0, stream>>>(cvf, cvt_b, T5C, T5C);
  cvt_pad_kernel<<<blocks((size_t)IMGPC * DIMC), 256, 0, stream>>>(ckif, cki_b, IMGPC, DIMC, IMGC, 1.0f);
  vtrans_kernel<<<blocks((size_t)HEADSC * HDC * IMGPC), 256, 0, stream>>>(cvif, cvit_b, IMGPC, IMGC);

  flash(q_b, ck_b, cvt_b, attn, T5C, T5C, 0);          // text attention (overwrite)
  flash(q_b, cki_b, cvit_b, attn, IMGC, IMGPC, 1);     // + image attention (accumulate)
  cvt_pad_kernel<<<blocks(SD), 256, 0, stream>>>(attn, attn_b, STOK, DIMC, STOK, 1.0f);
  gemm(attn_b, cw_oT, ca_ob, yb, nullptr, STOK, DIMC, DIMC, 0);
  resadd_kernel<<<blocks(SD), 256, 0, stream>>>(x1, yb, x2, (int)SD);

  // ---- FFN ----
  bf16* z_b = xs_b;         // reuse
  ln_kernel<<<STOK, 256, 0, stream>>>(x2, nullptr, z_b, e6 + 4 * DIMC, e6 + 3 * DIMC, 1.0f, DIMC);
  gemm(z_b, w1T, ffn_b1, nullptr, h_b, STOK, FFNC, DIMC, 1);   // fused tanh-GELU -> bf16
  gemm(h_b, w2T, ffn_b2, yb, nullptr, STOK, DIMC, FFNC, 0);
  resgate_kernel<<<blocks(SD), 256, 0, stream>>>(x2, yb, e6 + 5 * DIMC, out, (int)SD);
}
